// FCLSTM_54468775248344
// MI455X (gfx1250) — matrix operand order repaired and checked by reading, compiled, not run on hardware
//
#include <hip/hip_runtime.h>
#include <hip/hip_bf16.h>

// Problem constants (match reference)
constexpr int VOCAB = 32000;
constexpr int E = 512;     // EMBED
constexpr int H = 1024;    // HIDDEN
constexpr int NCLS = 2;
constexpr int B = 64;
constexpr int S = 512;

// Recurrent persistent kernel geometry: 64 blocks x 4 waves = 256 tiles,
// one workgroup per WGP (294.75 KB LDS of the 320 KB WGP pool).
constexpr int REC_BLOCKS = 64;
constexpr int REC_THREADS = 128;           // 4 waves
constexpr int PITCH = 1048;                // LDS row pitch in halfs: 524 dwords,
                                           // 524 % 64 = 12 -> conflict-free row starts
constexpr int LDS_HALFS = 2 * 64 * PITCH + 16 * PITCH;   // Wf + Wh + A
constexpr size_t LDS_BYTES = (size_t)LDS_HALFS * 2;      // 301,824 B <= 320 KB

typedef __attribute__((ext_vector_type(16))) _Float16 v16h;
typedef __attribute__((ext_vector_type(8)))  _Float16 v8h;
typedef __attribute__((ext_vector_type(8)))  float    v8f;
typedef __attribute__((ext_vector_type(4)))  float    v4f;

// ---------------------------------------------------------------------------
// CDNA5 async global->LDS staging (GLOBAL_LOAD_ASYNC_TO_LDS_B128, ASYNCcnt).
// Probe feedback: the builtin's pointer params are int __vector(4)* with
// address-space qualifiers.  Guarded: falls back to load+ds_store if absent.
// ---------------------------------------------------------------------------
#if __has_builtin(__builtin_amdgcn_global_load_async_to_lds_b128) && \
    __has_builtin(__builtin_amdgcn_s_wait_asynccnt)
#define USE_ASYNC_LDS 1
#else
#define USE_ASYNC_LDS 0
#endif

typedef int async_v4i __attribute__((vector_size(16)));
typedef __attribute__((address_space(1))) async_v4i* async_gptr;
typedef __attribute__((address_space(3))) async_v4i* async_lptr;

__device__ __forceinline__ void copy16_g2l(_Float16* ldst, const _Float16* gsrc) {
#if USE_ASYNC_LDS
  __builtin_amdgcn_global_load_async_to_lds_b128(
      (async_gptr)(void*)gsrc,
      (async_lptr)(void*)ldst,
      0, 0);
#else
  *(v8h*)ldst = *(const v8h*)gsrc;
#endif
}

__device__ __forceinline__ void staging_wait() {
#if USE_ASYNC_LDS
  __builtin_amdgcn_s_wait_asynccnt(0);
#endif
}

// ---------------------------------------------------------------------------
// WMMA helpers (CDNA5 wave32, V_WMMA_F32_16X16X32_F16)
// ---------------------------------------------------------------------------
__device__ __forceinline__ v8f wmma_f16(v16h a, v16h b, v8f c) {
  return __builtin_amdgcn_wmma_f32_16x16x32_f16(false, a, false, b, (short)0, c,
                                                false, false);
}

// A-matrix 16x32 f16 chunk.  p must already include: row*pitch + (laneHi? 8:0).
__device__ __forceinline__ v16h ldA(const _Float16* p) {
  v8h lo = *(const v8h*)(p);
  v8h hi = *(const v8h*)(p + 16);
  return __builtin_shufflevector(lo, hi, 0, 1, 2, 3, 4, 5, 6, 7,
                                 8, 9, 10, 11, 12, 13, 14, 15);
}

// B-matrix 32x16 f16 chunk from row-major W[j,k] (== B^T).
// p must already include: col*pitch + (laneHi? 8:0).
__device__ __forceinline__ v16h ldB(const _Float16* p) {
  v8h lo = *(const v8h*)(p);
  v8h hi = *(const v8h*)(p + 16);
  return __builtin_shufflevector(lo, hi, 0, 1, 2, 3, 4, 5, 6, 7,
                                 8, 9, 10, 11, 12, 13, 14, 15);
}

// ---------------------------------------------------------------------------
// Grid-wide barrier keyed to the step index (monotone generation counter, so
// no arrive/read race).  prep_kernel zeroes cnt/gen each launch (stream order).
// ---------------------------------------------------------------------------
__device__ __forceinline__ void grid_barrier(unsigned* cnt, unsigned* gen,
                                             unsigned nblk, unsigned it) {
  __syncthreads();
  if (threadIdx.x == 0) {
    __threadfence();
    if (atomicAdd(cnt, 1u) == nblk - 1u) {
      atomicExch(cnt, 0u);
      __threadfence();
      atomicAdd(gen, 1u);
    } else {
      volatile unsigned* vgen = gen;
      while (*vgen < it + 1u) { __builtin_amdgcn_s_sleep(2); }
    }
    __threadfence();
  }
  __syncthreads();
}

// ---------------------------------------------------------------------------
// Kernel 1: convert weights f32 -> f16, zero h0 and barrier state
// ---------------------------------------------------------------------------
__global__ void prep_kernel(const float* __restrict__ Wi, const float* __restrict__ Wf,
                            const float* __restrict__ Wh, const float* __restrict__ Wo,
                            _Float16* __restrict__ Wi16, _Float16* __restrict__ Wf16,
                            _Float16* __restrict__ Wh16, _Float16* __restrict__ Wo16,
                            _Float16* __restrict__ h0, unsigned* __restrict__ bar) {
  const int stride = blockDim.x * gridDim.x;
  const int i0 = blockIdx.x * blockDim.x + threadIdx.x;
  for (int i = i0; i < H * E; i += stride) Wi16[i] = (_Float16)Wi[i];
  for (int i = i0; i < H * H; i += stride) {
    Wf16[i] = (_Float16)Wf[i];
    Wh16[i] = (_Float16)Wh[i];
    Wo16[i] = (_Float16)Wo[i];
  }
  for (int i = i0; i < B * H; i += stride) h0[i] = (_Float16)0.0f;
  if (i0 < 2) bar[i0] = 0u;
}

// ---------------------------------------------------------------------------
// Kernel 2: G[t*B+b, j] = relu(emb[x[b,t]] @ W_i.T + b_i)   [32768 x 1024]
// One wave per 16x16 output tile; K=512 -> 16 WMMAs.
// ---------------------------------------------------------------------------
__global__ void input_proj_kernel(const int* __restrict__ x,
                                  const float* __restrict__ emb,
                                  const _Float16* __restrict__ Wi16,
                                  const float* __restrict__ b_i,
                                  _Float16* __restrict__ G) {
  const int lane = threadIdx.x & 31;
  const int wv = threadIdx.x >> 5;                 // 0..7
  const int mt = blockIdx.x >> 3;                  // 0..2047 (row tile of B*S)
  const int nt = ((blockIdx.x & 7) << 3) | wv;     // 0..63   (col tile of H)
  const int l15 = lane & 15;
  const int lh = lane >> 4;

  const int row = mt * 16 + l15;                   // global row = t*B + b
  const int t = row >> 6;                          // B == 64
  const int b = row & 63;
  const float* erow = emb + (size_t)x[b * S + t] * E + (lh ? 8 : 0);
  const _Float16* brow = Wi16 + (size_t)(nt * 16 + l15) * E + (lh ? 8 : 0);

  v8f c = {};
#pragma unroll 4
  for (int kb = 0; kb < E; kb += 32) {
    v4f a0 = *(const v4f*)(erow + kb);
    v4f a1 = *(const v4f*)(erow + kb + 4);
    v4f a2 = *(const v4f*)(erow + kb + 16);
    v4f a3 = *(const v4f*)(erow + kb + 20);
    v16h a;
#pragma unroll
    for (int i = 0; i < 4; ++i) {
      a[i]      = (_Float16)a0[i];
      a[4 + i]  = (_Float16)a1[i];
      a[8 + i]  = (_Float16)a2[i];
      a[12 + i] = (_Float16)a3[i];
    }
    c = wmma_f16(a, ldB(brow + kb), c);
  }

  const float bias = b_i[nt * 16 + l15];
  _Float16* gp = G + (size_t)(mt * 16 + lh * 8) * H + (nt * 16 + l15);
#pragma unroll
  for (int r = 0; r < 8; ++r) {
    float v = c[r] + bias;
    v = fmaxf(v, 0.0f);
    gp[(size_t)r * H] = (_Float16)v;
  }
}

// ---------------------------------------------------------------------------
// Kernel 3: persistent recurrent kernel, LDS-staged weights.
// 64 blocks x 128 threads (4 waves).  Block (mt = blk/16, ntg = blk%16) owns
// output strip rows [16mt,16mt+16) x cols [64ntg, 64ntg+64); wave w owns the
// 16x16 tile at cols 64ntg + 16w.
//   - W_f / W_h strips (2 x 64 x 1024 f16) async-staged to LDS ONCE.
//   - A tile (h rows, 32 KB) async-staged to LDS once per step.
//   - Inner loop: ds_load_b128 + v_wmma_f32_16x16x32_f16 only.
// LDS pitch 1048 halfs (524 dwords, 524%64==12) -> 16 lane rows start on
// distinct banks, every access 16B-aligned.
// ---------------------------------------------------------------------------
__global__ void recurrent_kernel(const _Float16* __restrict__ G,
                                 const _Float16* __restrict__ Wf16,
                                 const _Float16* __restrict__ Wh16,
                                 const _Float16* __restrict__ Wo16,
                                 const float* __restrict__ b_f,
                                 const float* __restrict__ b_h,
                                 const float* __restrict__ b_o,
                                 const int* __restrict__ lengths,
                                 _Float16* __restrict__ h0,
                                 _Float16* __restrict__ h1,
                                 _Float16* __restrict__ Hsel,
                                 float* __restrict__ lin,
                                 unsigned* __restrict__ bar) {
  extern __shared__ _Float16 lds_buf[];
  _Float16* lWf = lds_buf;                    // [64][PITCH]
  _Float16* lWh = lds_buf + 64 * PITCH;       // [64][PITCH]
  _Float16* lA  = lds_buf + 2 * 64 * PITCH;   // [16][PITCH]

  const int tid = threadIdx.x;
  const int lane = tid & 31;
  const int wave = tid >> 5;                  // 0..3
  const int mt = blockIdx.x >> 4;             // 0..3
  const int ntg = blockIdx.x & 15;            // 0..15
  const int nt = ntg * 4 + wave;              // 0..63
  const int l15 = lane & 15;
  const int lh = lane >> 4;

  const int jcol = nt * 16 + l15;
  const int rbase = mt * 16 + lh * 8;         // C/D row base for this lane

  // ---- one-time weight staging: rows j in [64*ntg, 64*ntg+64) ----
  {
    const int j0 = ntg * 64;
    for (int c = tid; c < 64 * 128; c += REC_THREADS) {
      const int jr = c >> 7;
      const int kc = (c & 127) << 3;
      copy16_g2l(lWf + jr * PITCH + kc, Wf16 + (size_t)(j0 + jr) * H + kc);
      copy16_g2l(lWh + jr * PITCH + kc, Wh16 + (size_t)(j0 + jr) * H + kc);
    }
  }

  const _Float16* aLrow = lA + l15 * PITCH + (lh ? 8 : 0);
  const _Float16* bfL = lWf + (wave * 16 + l15) * PITCH + (lh ? 8 : 0);
  const _Float16* bhL = lWh + (wave * 16 + l15) * PITCH + (lh ? 8 : 0);

  const float biasf = b_f[jcol];
  const float biash = b_h[jcol];

  int lenm1[8];
#pragma unroll
  for (int r = 0; r < 8; ++r) lenm1[r] = lengths[rbase + r] - 1;

  staging_wait();      // drain ASYNCcnt for this wave
  __syncthreads();     // weights staged, visible block-wide

  for (int t = 0; t < S; ++t) {
    const _Float16* hr = (t & 1) ? h1 : h0;
    _Float16* hw = (t & 1) ? h0 : h1;

    // ---- stage A tile (16 x 1024 halfs) for this block's mt ----
    {
      const _Float16* hsrc = hr + (size_t)(mt * 16) * H;
      for (int c = tid; c < 16 * 128; c += REC_THREADS) {
        const int rr = c >> 7;
        const int kc = (c & 127) << 3;
        copy16_g2l(lA + rr * PITCH + kc, hsrc + (size_t)rr * H + kc);
      }
    }
    staging_wait();
    __syncthreads();

    const _Float16* gcol = G + ((size_t)t * B + rbase) * H + jcol;
    // speculative prefetch of next step's G tile (global_prefetch_b8)
    __builtin_prefetch(gcol + (size_t)B * H, 0, 1);

    v8f cf = {};
    v8f ch = {};
#pragma unroll 4
    for (int kb = 0; kb < H; kb += 32) {
      v16h a = ldA(aLrow + kb);
      cf = wmma_f16(a, ldB(bfL + kb), cf);
      ch = wmma_f16(a, ldB(bhL + kb), ch);
    }

    _Float16* hwp = hw + (size_t)rbase * H + jcol;
    _Float16* hsp = Hsel + (size_t)rbase * H + jcol;
#pragma unroll
    for (int r = 0; r < 8; ++r) {
      const float af = cf[r] + biasf;
      const float ah = ch[r] + biash;
      const float sg = 1.0f / (1.0f + __expf(-af));                 // sigmoid
      const float th = 1.0f - 2.0f / (__expf(2.0f * ah) + 1.0f);    // tanh
      const float g = (float)gcol[(size_t)r * H];
      const float hn = sg + th * g;
      hwp[(size_t)r * H] = (_Float16)hn;
      if (lenm1[r] == t) hsp[(size_t)r * H] = (_Float16)hn;
    }

    grid_barrier(bar, bar + 1, gridDim.x, (unsigned)t);
  }

  // ---- Output projection: lin = Hsel @ W_o.T + b_o (one-time, from global) ----
  const _Float16* srow = Hsel + (size_t)(mt * 16 + l15) * H + (lh ? 8 : 0);
  const _Float16* borow = Wo16 + (size_t)jcol * H + (lh ? 8 : 0);
  v8f co = {};
#pragma unroll 4
  for (int kb = 0; kb < H; kb += 32) {
    v8h alo = *(const v8h*)(srow + kb);
    v8h ahi = *(const v8h*)(srow + kb + 16);
    v16h a = __builtin_shufflevector(alo, ahi, 0, 1, 2, 3, 4, 5, 6, 7,
                                     8, 9, 10, 11, 12, 13, 14, 15);
    v8h blo = *(const v8h*)(borow + kb);
    v8h bhi = *(const v8h*)(borow + kb + 16);
    v16h bm = __builtin_shufflevector(blo, bhi, 0, 1, 2, 3, 4, 5, 6, 7,
                                      8, 9, 10, 11, 12, 13, 14, 15);
    co = wmma_f16(a, bm, co);
  }
  const float biaso = b_o[jcol];
#pragma unroll
  for (int r = 0; r < 8; ++r) {
    lin[(size_t)(rbase + r) * H + jcol] = co[r] + biaso;
  }
}

// ---------------------------------------------------------------------------
// Kernel 4: logits + log_softmax.  One block per batch row, NCLS == 2.
// ---------------------------------------------------------------------------
__global__ void logits_kernel(const float* __restrict__ lin,
                              const float* __restrict__ W_lin,
                              const float* __restrict__ b_lin,
                              float* __restrict__ out) {
  const int b = blockIdx.x;
  const int tid = threadIdx.x;
  __shared__ float s0s[256];
  __shared__ float s1s[256];
  float s0 = 0.0f, s1 = 0.0f;
  for (int k = tid; k < H; k += 256) {
    const float v = lin[(size_t)b * H + k];
    s0 += v * W_lin[k];
    s1 += v * W_lin[H + k];
  }
  s0s[tid] = s0;
  s1s[tid] = s1;
  __syncthreads();
  for (int s = 128; s > 0; s >>= 1) {
    if (tid < s) {
      s0s[tid] += s0s[tid + s];
      s1s[tid] += s1s[tid + s];
    }
    __syncthreads();
  }
  if (tid == 0) {
    const float z0 = s0s[0] + b_lin[0];
    const float z1 = s1s[0] + b_lin[1];
    const float m = fmaxf(z0, z1);
    const float lse = m + __logf(__expf(z0 - m) + __expf(z1 - m));
    out[b * 2 + 0] = z0 - lse;
    out[b * 2 + 1] = z1 - lse;
  }
}

// ---------------------------------------------------------------------------
extern "C" void kernel_launch(void* const* d_in, const int* in_sizes, int n_in,
                              void* d_out, int out_size, void* d_ws, size_t ws_size,
                              hipStream_t stream) {
  const int* x = (const int*)d_in[0];
  const int* lengths = (const int*)d_in[1];
  const float* emb = (const float*)d_in[2];
  const float* W_i = (const float*)d_in[3];
  const float* b_i = (const float*)d_in[4];
  const float* W_f = (const float*)d_in[5];
  const float* b_f = (const float*)d_in[6];
  const float* W_h = (const float*)d_in[7];
  const float* b_h = (const float*)d_in[8];
  const float* W_o = (const float*)d_in[9];
  const float* b_o = (const float*)d_in[10];
  const float* W_lin = (const float*)d_in[11];
  const float* b_lin = (const float*)d_in[12];
  float* out = (float*)d_out;

  char* w = (char*)d_ws;
  size_t off = 0;
  auto take = [&](size_t bytes) -> void* {
    void* p = w + off;
    off += (bytes + 255) & ~(size_t)255;
    return p;
  };
  _Float16* Wi16 = (_Float16*)take((size_t)H * E * 2);
  _Float16* Wf16 = (_Float16*)take((size_t)H * H * 2);
  _Float16* Wh16 = (_Float16*)take((size_t)H * H * 2);
  _Float16* Wo16 = (_Float16*)take((size_t)H * H * 2);
  _Float16* G    = (_Float16*)take((size_t)S * B * H * 2);
  _Float16* h0   = (_Float16*)take((size_t)B * H * 2);
  _Float16* h1   = (_Float16*)take((size_t)B * H * 2);
  _Float16* Hsel = (_Float16*)take((size_t)B * H * 2);
  float*    lin  = (float*)take((size_t)B * H * 4);
  unsigned* bar  = (unsigned*)take(256);

  hipLaunchKernelGGL(prep_kernel, dim3(512), dim3(256), 0, stream,
                     W_i, W_f, W_h, W_o, Wi16, Wf16, Wh16, Wo16, h0, bar);

  // (B*S/16) row tiles x (H/16) col tiles, 8 waves (col tiles) per block
  hipLaunchKernelGGL(input_proj_kernel, dim3((B * S / 16) * (H / 16 / 8)),
                     dim3(256), 0, stream, x, emb, Wi16, b_i, G);

  hipLaunchKernelGGL(recurrent_kernel, dim3(REC_BLOCKS), dim3(REC_THREADS),
                     LDS_BYTES, stream,
                     G, Wf16, Wh16, Wo16, b_f, b_h, b_o, lengths,
                     h0, h1, Hsel, lin, bar);

  hipLaunchKernelGGL(logits_kernel, dim3(B), dim3(256), 0, stream,
                     lin, W_lin, b_lin, out);
}
